// PTV3_Embedding_214748364934
// MI455X (gfx1250) — compile-verified, matrix operand order checked by reading, not run on hardware
//
#include <hip/hip_runtime.h>
#include <hip/hip_bf16.h>
#include <math.h>

// Problem constants (match reference)
#define N_ROWS  400000
#define IN_C    128
#define EMBED_C 512
#define LN_EPS  1e-5f
#define NBLK    (N_ROWS / 16)   // 25000 row blocks of 16 rows
#define GRID    2500            // persistent blocks; 10 row blocks each

typedef __attribute__((ext_vector_type(16))) __bf16 v16bf;
typedef __attribute__((ext_vector_type(8)))  float  v8f;

static __device__ __forceinline__ unsigned pack2bf16(float a, float b) {
  unsigned short ua = __builtin_bit_cast(unsigned short, (__bf16)a);
  unsigned short ub = __builtin_bit_cast(unsigned short, (__bf16)b);
  return (unsigned)ua | ((unsigned)ub << 16);
}
static __device__ __forceinline__ float bf16_rn_f32(float a) {
  return (float)((__bf16)a);  // value of round-to-nearest bf16(a)
}
// Branchless tanh-form GELU: v * sigmoid(1.595769122*(v + 0.044715 v^3)).
// exp/rcp are TRANS ops (co-execute). Max |err| vs exact-erf GELU ~5e-4.
static __device__ __forceinline__ float gelu_fast(float v) {
  float s = v * v;
  float w = v * fmaf(s, -0.07135481627f, -1.595769122f);  // w = -z
  float e = __expf(w);
  return v * __builtin_amdgcn_rcpf(1.0f + e);
}

// ---------------------------------------------------------------------------
// Prepass 1: W[512][128] f32 -> WMMA B-operand register image, bf16 hi + lo
// planes in d_ws. Record (ntile*4 + kchunk)*32 + lane = 16 bf16 (32 B) = one
// lane's v16bf. Element e -> K = kchunk*32 + (e<8 ? (lane<16?0:8)+e
//                                                 : (lane<16?16:24)+(e-8)).
// ---------------------------------------------------------------------------
__global__ __launch_bounds__(256) void prep_w_kernel(
    const float* __restrict__ W, v16bf* __restrict__ whi,
    v16bf* __restrict__ wlo) {
  int tid = blockIdx.x * blockDim.x + threadIdx.x;
  if (tid >= 32 * 4 * 32) return;
  int lane = tid & 31;
  int c    = (tid >> 5) & 3;
  int t    = tid >> 7;
  int n    = t * 16 + (lane & 15);
  int klo  = (lane < 16) ? 0 : 8;
  int khi  = (lane < 16) ? 16 : 24;
  v16bf hi, lo;
#pragma unroll
  for (int e = 0; e < 16; ++e) {
    int k   = c * 32 + (e < 8 ? klo + e : khi + (e - 8));
    float w = W[n * IN_C + k];
    __bf16 h = (__bf16)w;
    hi[e] = h;
    lo[e] = (__bf16)(w - (float)h);
  }
  whi[tid] = hi;
  wlo[tid] = lo;
}

// ---------------------------------------------------------------------------
// Prepass 2: w1[k] = sum_n W[n][k]  (k=0..127), w1[128] = sum_n b[n].
// Row means then come from x[m,:].w1 + Sb -- no post-GEMM sum reduction.
// ---------------------------------------------------------------------------
__global__ __launch_bounds__(256) void prep_sums_kernel(
    const float* __restrict__ W, const float* __restrict__ b,
    float* __restrict__ w1) {
  int tid = blockIdx.x * blockDim.x + threadIdx.x;
  if (tid < IN_C) {
    float s = 0.f;
    for (int n = 0; n < EMBED_C; ++n) s += W[n * IN_C + tid];
    w1[tid] = s;
  } else if (tid == IN_C) {
    float s = 0.f;
    for (int n = 0; n < EMBED_C; ++n) s += b[n];
    w1[IN_C] = s;
  }
}

// ---------------------------------------------------------------------------
// Persistent fused GEMM(bf16x2 split) + bias + LayerNorm + GELU.
// 512 threads = 16 waves; wave w owns cols [w*32, w*32+32): B hi+lo register-
// resident (128 VGPRs, loaded once). Per 16-row block:
//   A: zero LN stat buffers (parity double-buffered)        [barrier]
//   B: stage x block -> LDS bf16 hi/lo A-image (1 float4/thread) and fold
//      the per-row mean dot (x . w1) into LDS atomics        [barrier]
//   C: GEMM from LDS A + register B (24 WMMA), bias, sum(y^2) xor-tree +
//      ds atomics                                            [barrier]
//   D: mu/rstd, normalize, GELU, store.
// A-image is single-buffered (reads complete before barrier C; next writes
// happen after it). Stats arrays are double-buffered by iteration parity.
// ---------------------------------------------------------------------------
__global__ __launch_bounds__(512) void gemm_ln_gelu_kernel(
    const float* __restrict__ x, const float* __restrict__ b,
    const float* __restrict__ gamma, const float* __restrict__ beta,
    const v16bf* __restrict__ whi, const v16bf* __restrict__ wlo,
    const float* __restrict__ w1, float* __restrict__ out) {
  __shared__ __align__(16) unsigned a_hi[128 * 8];  // 4 KB A-image, hi plane
  __shared__ __align__(16) unsigned a_lo[128 * 8];  // 4 KB A-image, lo plane
  __shared__ float ln_dot[2][16];
  __shared__ float ln_sq[2][16];

  const int tid  = threadIdx.x;
  const int wave = tid >> 5;   // 0..15
  const int lane = tid & 31;
  const int nlan = lane & 15;
  const int radd = (lane < 16) ? 0 : 8;

  // staging map: one float4 of x per thread per row block
  const int rec   = tid >> 2;   // 0..127 = kchunk*32 + record-lane
  const int sc    = rec >> 5;   // k-chunk
  const int sl    = rec & 31;   // record lane
  const int epart = tid & 3;
  const int k0    = sc * 32 + (epart < 2
                        ? ((sl < 16) ? 0 : 8) + epart * 4
                        : ((sl < 16) ? 16 : 24) + (epart - 2) * 4);
  const int srow  = sl & 15;    // row within block
  const int di    = rec * 8 + epart * 2;  // dword index into A-image
  const float4 w1v = *(const float4*)(w1 + k0);
  const float  Sb  = w1[IN_C];

  // ---- stationary B: 2 N-tiles x 4 K-chunks, hi+lo (128 VGPRs) ----
  v16bf Bh[2][4], Bl[2][4];
#pragma unroll
  for (int tt = 0; tt < 2; ++tt) {
#pragma unroll
    for (int c = 0; c < 4; ++c) {
      const int idx = (((wave * 2 + tt) * 4) + c) * 32 + lane;
      Bh[tt][c] = whi[idx];
      Bl[tt][c] = wlo[idx];
    }
  }

  // ---- loop-invariant epilogue params for this wave's 32 columns ----
  float bv[2], g[2], bt[2];
#pragma unroll
  for (int tt = 0; tt < 2; ++tt) {
    const int n = wave * 32 + tt * 16 + nlan;
    bv[tt] = b[n];
    g[tt]  = gamma[n];
    bt[tt] = beta[n];
  }

  int it = 0;
  for (int blk = blockIdx.x; blk < NBLK; blk += GRID, ++it) {
    const int m0  = blk * 16;
    const int buf = it & 1;

    // ---- step A: zero this iteration's stat buffers ----
    if (tid < 16) {
      ln_dot[buf][tid] = 0.f;
      ln_sq[buf][tid]  = 0.f;
    }
    __syncthreads();

    // ---- step B: stage A block to LDS (bf16 hi/lo) + mean dot ----
    {
      const float* xp = x + (size_t)(m0 + srow) * IN_C + k0;
      float4 q = *(const float4*)xp;
      if (blk + GRID < NBLK)  // gfx1250 global_prefetch of next row block
        __builtin_prefetch(xp + (size_t)GRID * 16 * IN_C, 0, 1);
      unsigned h0 = pack2bf16(q.x, q.y);
      unsigned h1 = pack2bf16(q.z, q.w);
      float r0 = q.x - bf16_rn_f32(q.x);
      float r1 = q.y - bf16_rn_f32(q.y);
      float r2 = q.z - bf16_rn_f32(q.z);
      float r3 = q.w - bf16_rn_f32(q.w);
      unsigned l0 = pack2bf16(r0, r1);
      unsigned l1 = pack2bf16(r2, r3);
      *(uint2*)&a_hi[di] = make_uint2(h0, h1);
      *(uint2*)&a_lo[di] = make_uint2(l0, l1);
      float pd = q.x * w1v.x + q.y * w1v.y + q.z * w1v.z + q.w * w1v.w;
      atomicAdd(&ln_dot[buf][srow], pd);
    }
    __syncthreads();

    // ---- step C: GEMM from LDS A-image + register B ----
    v8f acc0 = {}, acc1 = {};
#pragma unroll
    for (int c = 0; c < 4; ++c) {
      v16bf ah = *(const v16bf*)&a_hi[(c * 32 + lane) * 8];
      v16bf al = *(const v16bf*)&a_lo[(c * 32 + lane) * 8];
      acc0 = __builtin_amdgcn_wmma_f32_16x16x32_bf16(
          false, ah, false, Bh[0][c], (short)0, acc0, false, false);
      acc1 = __builtin_amdgcn_wmma_f32_16x16x32_bf16(
          false, ah, false, Bh[1][c], (short)0, acc1, false, false);
      acc0 = __builtin_amdgcn_wmma_f32_16x16x32_bf16(
          false, al, false, Bh[0][c], (short)0, acc0, false, false);
      acc1 = __builtin_amdgcn_wmma_f32_16x16x32_bf16(
          false, al, false, Bh[1][c], (short)0, acc1, false, false);
      acc0 = __builtin_amdgcn_wmma_f32_16x16x32_bf16(
          false, ah, false, Bl[0][c], (short)0, acc0, false, false);
      acc1 = __builtin_amdgcn_wmma_f32_16x16x32_bf16(
          false, ah, false, Bl[1][c], (short)0, acc1, false, false);
    }

    // bias + per-row sum(y^2) partial over this wave's 32 cols
    float sq[8];
#pragma unroll
    for (int r = 0; r < 8; ++r) {
      float y0 = acc0[r] + bv[0];
      float y1 = acc1[r] + bv[1];
      acc0[r] = y0;
      acc1[r] = y1;
      sq[r] = fmaf(y0, y0, y1 * y1);
    }
    // reduce over the 16 N-lanes (xor tree stays within 16-lane halves)
#pragma unroll
    for (int r = 0; r < 8; ++r) {
      sq[r] += __shfl_xor(sq[r], 1, 32);
      sq[r] += __shfl_xor(sq[r], 2, 32);
      sq[r] += __shfl_xor(sq[r], 4, 32);
      sq[r] += __shfl_xor(sq[r], 8, 32);
    }
    if (lane == 0 || lane == 16) {
      const int rb = (lane == 16) ? 8 : 0;
#pragma unroll
      for (int r = 0; r < 8; ++r) atomicAdd(&ln_sq[buf][rb + r], sq[r]);
    }
    __syncthreads();

    // ---- step D: mu/rstd + normalize + GELU + store ----
    float mu[8], rstd[8];
#pragma unroll
    for (int r = 0; r < 8; ++r) {
      float mn  = (ln_dot[buf][r + radd] + Sb) * (1.0f / EMBED_C);
      float var = ln_sq[buf][r + radd] * (1.0f / EMBED_C) - mn * mn;
      mu[r]   = mn;
      rstd[r] = rsqrtf(var + LN_EPS);
    }
#pragma unroll
    for (int tt = 0; tt < 2; ++tt) {
      const int n = wave * 32 + tt * 16 + nlan;
#pragma unroll
      for (int r = 0; r < 8; ++r) {
        float y = (tt == 0) ? acc0[r] : acc1[r];
        float v = fmaf((y - mu[r]) * rstd[r], g[tt], bt[tt]);
        const int row = m0 + r + radd;
        out[(size_t)row * EMBED_C + n] = gelu_fast(v);
      }
    }
  }
}

// ---------------------------------------------------------------------------
// Launch: two tiny prepasses (B images 256 KB + w1/Sb 516 B in d_ws), then
// the persistent fused kernel. Inputs x, W, b, gamma, beta (f32). Out f32.
// ---------------------------------------------------------------------------
extern "C" void kernel_launch(void* const* d_in, const int* in_sizes, int n_in,
                              void* d_out, int out_size, void* d_ws,
                              size_t ws_size, hipStream_t stream) {
  const float* x     = (const float*)d_in[0];
  const float* W     = (const float*)d_in[1];
  const float* b     = (const float*)d_in[2];
  const float* gamma = (const float*)d_in[3];
  const float* beta  = (const float*)d_in[4];
  float* out = (float*)d_out;

  v16bf* whi = (v16bf*)d_ws;          // 128 KB
  v16bf* wlo = whi + 32 * 4 * 32;     // 128 KB
  float* w1  = (float*)((char*)d_ws + 2 * 32 * 4 * 32 * 32);  // 129 floats

  prep_w_kernel<<<16, 256, 0, stream>>>(W, whi, wlo);
  prep_sums_kernel<<<1, 256, 0, stream>>>(W, b, w1);
  gemm_ln_gelu_kernel<<<GRID, 512, 0, stream>>>(x, b, gamma, beta, whi, wlo,
                                                w1, out);
}
